// CausalSelfAttention_51634096833360
// MI455X (gfx1250) — matrix operand order repaired and checked by reading, not compiled, not run on hardware
//
#include <hip/hip_runtime.h>
#include <hip/hip_bf16.h>

typedef __attribute__((ext_vector_type(16))) __bf16 v16bf;
typedef __attribute__((ext_vector_type(8)))  float  v8f;

union FragBF { v16bf v; unsigned u[8]; };

__device__ __forceinline__ unsigned short f2bf(float f) {
  unsigned u = __float_as_uint(f);
  u += 0x7FFFu + ((u >> 16) & 1u);   // round-to-nearest-even
  return (unsigned short)(u >> 16);
}

// ---------------------------------------------------------------- convert
__global__ __launch_bounds__(256)
void cvt_f32_to_bf16(const float* __restrict__ in, unsigned short* __restrict__ out, int n) {
  int i = blockIdx.x * 256 + threadIdx.x;
  if (i < n) out[i] = f2bf(in[i]);
}

// ---------------------------------------------------------------- GEMM
// C[M,N] = A[M,K] (bf16) * W[K,N] (bf16) + bias, tile 128x64, BK=32, 8 waves.
// Double-buffered pipeline: A tile via GLOBAL_LOAD_ASYNC_TO_LDS_B128 (ASYNCcnt),
// W tile via VGPR-staged transpose store; one barrier per K-step.
// mode 0: store bf16 [B,H,T,Dh]   (Q,K)
// mode 1: store bf16 [B,H,Dh,T]   (V transposed)
// mode 2: store fp32 [M,N]        (final projection)
#define BM 128
#define BN 64
#define BK 32
#define LDSW 34   // padded LDS row stride (bf16 elems); 17 dwords -> odd stride

__global__ __launch_bounds__(256)
void gemm_wmma_bf16(const unsigned short* __restrict__ A,
                    const unsigned short* __restrict__ W,
                    const float* __restrict__ bias,
                    void* __restrict__ out,
                    int M, int N, int K, int mode, int T, int H) {
  __shared__ unsigned short As[2][BM * LDSW];
  __shared__ unsigned short Bs[2][BN * LDSW];

  const int tid  = threadIdx.x;
  const int wave = tid >> 5, lane = tid & 31;
  const int hlf  = lane >> 4, l16 = lane & 15;
  const int bm = blockIdx.x * BM, bn = blockIdx.y * BN;

  v8f c[4] = {};
  const int arow = tid >> 1, acol = (tid & 1) * 16;

  // async copy of this thread's 32B slice of the 128x32 A tile into As[buf]
  auto issueA = [&](int buf, int kk) {
    unsigned lds0 = (unsigned)(size_t)(&As[buf][arow * LDSW + acol]);
    unsigned long long g0 =
        (unsigned long long)(const void*)(A + (size_t)(bm + arow) * K + kk + acol);
    asm volatile("global_load_async_to_lds_b128 %0, %1, off"
                 :: "v"(lds0), "v"(g0) : "memory");
    asm volatile("global_load_async_to_lds_b128 %0, %1, off offset:16"
                 :: "v"(lds0), "v"(g0) : "memory");
  };

  unsigned short wreg[8];
  auto loadW = [&](int kk) {
    #pragma unroll
    for (int i = 0; i < 8; ++i) {
      int e = tid * 8 + i;
      int kq = e >> 6, n = e & 63;
      wreg[i] = W[(size_t)(kk + kq) * N + bn + n];
    }
  };
  auto storeW = [&](int buf) {        // transposed: Bs[n][k] so k-pairs are contiguous
    #pragma unroll
    for (int i = 0; i < 8; ++i) {
      int e = tid * 8 + i;
      int kq = e >> 6, n = e & 63;
      Bs[buf][n * LDSW + kq] = wreg[i];
    }
  };

  // prologue: fill buffer 0
  issueA(0, 0);
  loadW(0);
  storeW(0);

  const int steps = K / BK;
  int buf = 0;
  for (int s = 0; s < steps; ++s) {
    asm volatile("s_wait_asynccnt 0x0" ::: "memory");  // own async A writes done
    __syncthreads();                                   // all waves' tiles visible

    const int nxt = buf ^ 1;
    const bool more = (s + 1 < steps);
    if (more) {                     // overlap next tile fetch with this tile's WMMAs
      issueA(nxt, (s + 1) * BK);
      loadW((s + 1) * BK);
    }

    FragBF a;
    #pragma unroll
    for (int v = 0; v < 8; ++v) {
      int ka = ((v < 4) ? 0 : 16) + (hlf ? 8 : 0) + 2 * (v & 3);
      a.u[v] = *(const unsigned*)(&As[buf][(wave * 16 + l16) * LDSW + ka]);
    }
    #pragma unroll
    for (int nt = 0; nt < 4; ++nt) {
      FragBF b;
      #pragma unroll
      for (int v = 0; v < 8; ++v) {
        int kb = ((v < 4) ? 0 : 16) + (hlf ? 8 : 0) + 2 * (v & 3);
        b.u[v] = *(const unsigned*)(&Bs[buf][(nt * 16 + l16) * LDSW + kb]);
      }
      c[nt] = __builtin_amdgcn_wmma_f32_16x16x32_bf16(
          false, a.v, false, b.v, (short)0, c[nt], false, false);
    }

    if (more) storeW(nxt);          // other buffer; next barrier covers visibility
    buf = nxt;
  }

  // epilogue
  #pragma unroll
  for (int nt = 0; nt < 4; ++nt) {
    int col = bn + nt * 16 + l16;
    float bsv = bias[col];
    #pragma unroll
    for (int r = 0; r < 8; ++r) {
      int row = bm + wave * 16 + r + hlf * 8;
      float val = c[nt][r] + bsv;
      if (mode == 2) {
        ((float*)out)[(size_t)row * N + col] = val;
      } else {
        int b_ = row / T, t = row - b_ * T;
        int h = col >> 6, d = col & 63;
        unsigned short bf = f2bf(val);
        if (mode == 0)
          ((unsigned short*)out)[(((size_t)(b_ * H + h) * T + t) << 6) + d] = bf;
        else
          ((unsigned short*)out)[(((size_t)(b_ * H + h) << 6) + d) * T + t] = bf;
      }
    }
  }
}

// ---------------------------------------------------------------- attention
// One wave per 16-query tile; streams 32 keys/step with online softmax.
// q,k: bf16 [B,H,T,64]; vt: bf16 [B,H,64,T]; y: bf16 [B,T,H*64]
__global__ __launch_bounds__(256)
void attn_flash_wmma(const unsigned short* __restrict__ q,
                     const unsigned short* __restrict__ k,
                     const unsigned short* __restrict__ vt,
                     unsigned short* __restrict__ y,
                     int H, int T, float scale) {
  __shared__ unsigned short Pl[8][16 * 32];   // per-wave P staging (C-layout -> A-layout)

  const int tid  = threadIdx.x;
  const int wave = tid >> 5, lane = tid & 31;
  const int hlf  = lane >> 4, l16 = lane & 15;
  const int tilesPerBH = T >> 4;
  const int gid = blockIdx.x * 8 + wave;
  const int bh  = gid / tilesPerBH;
  const int qt  = gid - bh * tilesPerBH;
  const int q0  = qt << 4;

  const unsigned short* qb = q  + (size_t)bh * T * 64 + (size_t)q0 * 64;
  const unsigned short* kb = k  + (size_t)bh * T * 64;
  const unsigned short* vb = vt + (size_t)bh * 64 * T;
  unsigned short* P = &Pl[wave][0];

  FragBF qa0, qa1;
  #pragma unroll
  for (int v = 0; v < 8; ++v) {
    int ka = ((v < 4) ? 0 : 16) + (hlf ? 8 : 0) + 2 * (v & 3);
    qa0.u[v] = *(const unsigned*)(qb + l16 * 64 + ka);
    qa1.u[v] = *(const unsigned*)(qb + l16 * 64 + 32 + ka);
  }

  v8f o[4] = {};
  float mr[8], lr[8];
  #pragma unroll
  for (int r = 0; r < 8; ++r) { mr[r] = -3.0e38f; lr[r] = 0.0f; }

  for (int k0 = 0; k0 < q0 + 16; k0 += 32) {   // causal: never visits fully-masked steps
    v8f s0 = {}, s1 = {};
    FragBF bA, bB;
    #pragma unroll
    for (int v = 0; v < 8; ++v) {
      int dk = ((v < 4) ? 0 : 16) + (hlf ? 8 : 0) + 2 * (v & 3);
      bA.u[v] = *(const unsigned*)(kb + (size_t)(k0 + l16) * 64 + dk);
      bB.u[v] = *(const unsigned*)(kb + (size_t)(k0 + 16 + l16) * 64 + dk);
    }
    s0 = __builtin_amdgcn_wmma_f32_16x16x32_bf16(false, qa0.v, false, bA.v, (short)0, s0, false, false);
    s1 = __builtin_amdgcn_wmma_f32_16x16x32_bf16(false, qa0.v, false, bB.v, (short)0, s1, false, false);
    #pragma unroll
    for (int v = 0; v < 8; ++v) {
      int dk = 32 + ((v < 4) ? 0 : 16) + (hlf ? 8 : 0) + 2 * (v & 3);
      bA.u[v] = *(const unsigned*)(kb + (size_t)(k0 + l16) * 64 + dk);
      bB.u[v] = *(const unsigned*)(kb + (size_t)(k0 + 16 + l16) * 64 + dk);
    }
    s0 = __builtin_amdgcn_wmma_f32_16x16x32_bf16(false, qa1.v, false, bA.v, (short)0, s0, false, false);
    s1 = __builtin_amdgcn_wmma_f32_16x16x32_bf16(false, qa1.v, false, bB.v, (short)0, s1, false, false);

    // online softmax over this 16x32 score tile (rows reduced across 16 lanes)
    #pragma unroll
    for (int r = 0; r < 8; ++r) {
      int query = q0 + r + hlf * 8;
      float x0 = s0[r] * scale; if (k0 + l16 > query)      x0 = -3.0e38f;
      float x1 = s1[r] * scale; if (k0 + 16 + l16 > query) x1 = -3.0e38f;
      float tm = fmaxf(x0, x1);
      tm = fmaxf(tm, __shfl_xor(tm, 1));
      tm = fmaxf(tm, __shfl_xor(tm, 2));
      tm = fmaxf(tm, __shfl_xor(tm, 4));
      tm = fmaxf(tm, __shfl_xor(tm, 8));
      float mnew  = fmaxf(mr[r], tm);
      float alpha = __expf(mr[r] - mnew);
      float p0 = __expf(x0 - mnew);
      float p1 = __expf(x1 - mnew);
      float sum = p0 + p1;
      sum += __shfl_xor(sum, 1);
      sum += __shfl_xor(sum, 2);
      sum += __shfl_xor(sum, 4);
      sum += __shfl_xor(sum, 8);
      lr[r] = lr[r] * alpha + sum;
      mr[r] = mnew;
      int m = r + hlf * 8;
      P[m * 32 + l16]      = f2bf(p0);
      P[m * 32 + 16 + l16] = f2bf(p1);
      o[0][r] *= alpha; o[1][r] *= alpha; o[2][r] *= alpha; o[3][r] *= alpha;
    }

    // P (C-layout) -> A-fragment via per-wave LDS patch (same-wave DS ops are in-order)
    FragBF pa;
    #pragma unroll
    for (int v = 0; v < 8; ++v) {
      int ka = ((v < 4) ? 0 : 16) + (hlf ? 8 : 0) + 2 * (v & 3);
      pa.u[v] = *(const unsigned*)(P + l16 * 32 + ka);
    }
    #pragma unroll
    for (int nt = 0; nt < 4; ++nt) {
      FragBF bv_;
      #pragma unroll
      for (int v = 0; v < 8; ++v) {
        int key = k0 + ((v < 4) ? 0 : 16) + (hlf ? 8 : 0) + 2 * (v & 3);
        bv_.u[v] = *(const unsigned*)(vb + (size_t)(nt * 16 + l16) * T + key);
      }
      o[nt] = __builtin_amdgcn_wmma_f32_16x16x32_bf16(
          false, pa.v, false, bv_.v, (short)0, o[nt], false, false);
    }
  }

  // epilogue: y[b, t, h*64+d] bf16 (merged heads, ready as next GEMM's A)
  int b_ = bh / H, h = bh - b_ * H;
  #pragma unroll
  for (int nt = 0; nt < 4; ++nt) {
    #pragma unroll
    for (int r = 0; r < 8; ++r) {
      int tq  = q0 + r + hlf * 8;
      int col = (h << 6) + nt * 16 + l16;
      float val = o[nt][r] / lr[r];
      y[((size_t)b_ * T + tq) * (size_t)(H << 6) + col] = f2bf(val);
    }
  }
}

// ---------------------------------------------------------------- launcher
extern "C" void kernel_launch(void* const* d_in, const int* in_sizes, int n_in,
                              void* d_out, int out_size, void* d_ws, size_t ws_size,
                              hipStream_t stream) {
  const float* x  = (const float*)d_in[0];
  const float* Wq = (const float*)d_in[1];
  const float* bq = (const float*)d_in[2];
  const float* Wk = (const float*)d_in[3];
  const float* bk = (const float*)d_in[4];
  const float* Wv = (const float*)d_in[5];
  const float* bv = (const float*)d_in[6];
  const float* Wp = (const float*)d_in[7];
  const float* bp = (const float*)d_in[8];

  const int B = 2, T = 2048, C = 1024, H = 16;
  const int M = B * T;                    // 4096
  const float scale = 0.125f;             // 1/sqrt(64)

  char* ws = (char*)d_ws;
  const size_t MB = 1024ull * 1024ull;
  unsigned short* xb  = (unsigned short*)(ws);              //  8 MB: x bf16
  unsigned short* wqb = (unsigned short*)(ws +  8 * MB);    //  2 MB
  unsigned short* wkb = (unsigned short*)(ws + 10 * MB);    //  2 MB
  unsigned short* wvb = (unsigned short*)(ws + 12 * MB);    //  2 MB
  unsigned short* wpb = (unsigned short*)(ws + 14 * MB);    //  2 MB
  unsigned short* qd  = (unsigned short*)(ws + 16 * MB);    //  8 MB [B,H,T,Dh]
  unsigned short* kd  = (unsigned short*)(ws + 24 * MB);    //  8 MB [B,H,T,Dh]
  unsigned short* vtd = (unsigned short*)(ws + 32 * MB);    //  8 MB [B,H,Dh,T]
  unsigned short* yb  = (unsigned short*)(ws + 40 * MB);    //  8 MB [B,T,C]

  cvt_f32_to_bf16<<<(M * C + 255) / 256, 256, 0, stream>>>(x,  xb,  M * C);
  cvt_f32_to_bf16<<<(C * C + 255) / 256, 256, 0, stream>>>(Wq, wqb, C * C);
  cvt_f32_to_bf16<<<(C * C + 255) / 256, 256, 0, stream>>>(Wk, wkb, C * C);
  cvt_f32_to_bf16<<<(C * C + 255) / 256, 256, 0, stream>>>(Wv, wvb, C * C);
  cvt_f32_to_bf16<<<(C * C + 255) / 256, 256, 0, stream>>>(Wp, wpb, C * C);

  dim3 gg(M / BM, C / BN);
  gemm_wmma_bf16<<<gg, 256, 0, stream>>>(xb, wqb, bq, qd,    M, C, C, 0, T, H);
  gemm_wmma_bf16<<<gg, 256, 0, stream>>>(xb, wkb, bk, kd,    M, C, C, 0, T, H);
  gemm_wmma_bf16<<<gg, 256, 0, stream>>>(xb, wvb, bv, vtd,   M, C, C, 1, T, H);

  attn_flash_wmma<<<(B * H * (T / 16)) / 8, 256, 0, stream>>>(qd, kd, vtd, yb, H, T, scale);

  gemm_wmma_bf16<<<gg, 256, 0, stream>>>(yb, wpb, bp, d_out, M, C, C, 2, T, H);
}
